// LatentComposition_38671885534131
// MI455X (gfx1250) — matrix operand order repaired and checked by reading, not compiled, not run on hardware
//
#include <hip/hip_runtime.h>
#include <hip/hip_bf16.h>

#define NB   256   // batch
#define LSEQ 16    // sequence length
#define HH   512   // hidden
#define DD2  1024  // 2*H
#define GG4  2048  // 4*H (lstm gate width)
#define KLL  1024  // fused lstm K = H(x) + H(h)
#define NCC  5120  // 5*D2 (tree gate width)
#define KCC  2048  // 2*D2 (tree K)

typedef __attribute__((ext_vector_type(16))) __bf16 v16bf;
typedef __attribute__((ext_vector_type(8)))  float  v8f;

union Frag16 { v16bf v; unsigned int u[8]; };

__device__ __forceinline__ unsigned short f2bf(float f) {
    unsigned int u = __float_as_uint(f);
    u += 0x7FFFu + ((u >> 16) & 1u);          // round-to-nearest-even
    return (unsigned short)(u >> 16);
}
__device__ __forceinline__ float sigf(float x) { return 1.0f / (1.0f + __expf(-x)); }

// ---------------- prep kernels ----------------

__global__ void prep_x(const float* __restrict__ x, const int* __restrict__ len,
                       unsigned short* __restrict__ xf, unsigned short* __restrict__ xb)
{
    int i = blockIdx.x * blockDim.x + threadIdx.x;
    if (i >= NB * LSEQ * HH) return;
    int h = i % HH, t = (i / HH) % LSEQ, b = i / (HH * LSEQ);
    xf[i] = f2bf(x[i]);
    int ln = len[b];
    int st = (t < ln) ? (ln - 1 - t) : t;
    xb[i] = f2bf(x[(b * LSEQ + st) * HH + h]);
}

__global__ void prep_wcat(const float* __restrict__ wih, const float* __restrict__ whh,
                          unsigned short* __restrict__ wcat)
{
    int i = blockIdx.x * blockDim.x + threadIdx.x;
    if (i >= GG4 * KLL) return;
    int k = i % KLL, n = i / KLL;
    float v = (k < HH) ? wih[n * HH + k] : whh[n * HH + (k - HH)];
    wcat[i] = f2bf(v);
}

__global__ void prep_bf16(const float* __restrict__ src, unsigned short* __restrict__ dst, int n)
{
    int i = blockIdx.x * blockDim.x + threadIdx.x;
    if (i < n) dst[i] = f2bf(src[i]);
}

__global__ void zero_u16(unsigned short* __restrict__ p, int n)
{
    int i = blockIdx.x * blockDim.x + threadIdx.x;
    if (i < n) p[i] = 0;
}

// Fragment K maps (ISA 7.12.2):
//   A (16x32 bf16): pair p -> K = (p&4)*4 + hf*8 + (p&3)*2
//   B (32x16 bf16): pair p -> K = (p&4)*4 + hf*8 + (p&3)*2
__device__ __forceinline__ void load_afrag(Frag16& a, const unsigned short* __restrict__ base,
                                           int kk, int hf)
{
#pragma unroll
    for (int p = 0; p < 8; ++p) {
        int ka = kk + (((p & 4) << 2) | (hf << 3) | ((p & 3) << 1));
        a.u[p] = *(const unsigned int*)(base + ka);
    }
}
__device__ __forceinline__ void load_bfrag(Frag16& b, const unsigned short* __restrict__ base,
                                           int kk, int hf)
{
#pragma unroll
    for (int p = 0; p < 8; ++p) {
        int kb = kk + (((p & 4) << 2) | (hf << 3) | ((p & 3) << 1));
        b.u[p] = *(const unsigned int*)(base + kb);
    }
}

// ---------------- LSTM step GEMM: g = [x_t, h_{t-1}] @ wcat^T ----------------
// Register-blocked 64x32 macro-tile per wave (4x2 WMMA tiles).
// K loop split at the x/h boundary: no per-load pointer selects.
__global__ void lstm_step_gemm(const unsigned short* __restrict__ x16f,
                               const unsigned short* __restrict__ x16b,
                               const unsigned short* __restrict__ wcatf,
                               const unsigned short* __restrict__ wcatb,
                               const unsigned short* __restrict__ hstep16,
                               float* __restrict__ g, int t)
{
    const int dir = blockIdx.y;
    const unsigned short* x16 = dir ? x16b : x16f;
    const unsigned short* wc  = dir ? wcatb : wcatf;
    const unsigned short* hp  = hstep16 + dir * (NB * HH);
    float* go = g + dir * (NB * GG4);

    const int wave = threadIdx.x >> 5;
    const int lane = threadIdx.x & 31;
    const int idx  = blockIdx.x * 4 + wave;
    const int NMAC = (GG4 / 16) / 2;           // 64 macro-cols
    const int MMAC = (NB / 16) / 4;            // 4  macro-rows
    if (idx >= MMAC * NMAC) return;            // wave-uniform guard
    const int tm = idx / NMAC, tn = idx % NMAC;
    const int lr = lane & 15, hf = lane >> 4;

    const unsigned short* xrow[4];
    const unsigned short* hrow[4];
#pragma unroll
    for (int s = 0; s < 4; ++s) {
        int row = tm * 64 + s * 16 + lr;       // batch row
        xrow[s] = x16 + (row * LSEQ + t) * HH;
        hrow[s] = hp + row * HH;
    }
    const unsigned short* wrow[2];
    int col[2];
#pragma unroll
    for (int s = 0; s < 2; ++s) {
        col[s]  = tn * 32 + s * 16 + lr;
        wrow[s] = wc + col[s] * KLL;
    }

    v8f acc[4][2] = {};
    // --- first half: A from x_t, B cols [0,512) of K ---
#pragma unroll 2
    for (int kk = 0; kk < HH; kk += 32) {
        Frag16 a[4], b[2];
#pragma unroll
        for (int s = 0; s < 4; ++s) load_afrag(a[s], xrow[s], kk, hf);
#pragma unroll
        for (int s = 0; s < 2; ++s) load_bfrag(b[s], wrow[s], kk, hf);
#pragma unroll
        for (int s = 0; s < 4; ++s)
#pragma unroll
            for (int s2 = 0; s2 < 2; ++s2)
                acc[s][s2] = __builtin_amdgcn_wmma_f32_16x16x32_bf16(
                    false, a[s].v, false, b[s2].v, (short)0, acc[s][s2], false, false);
    }
    // --- second half: A from h_{t-1}, B cols [512,1024) of K ---
#pragma unroll 2
    for (int kk = 0; kk < HH; kk += 32) {
        Frag16 a[4], b[2];
#pragma unroll
        for (int s = 0; s < 4; ++s) load_afrag(a[s], hrow[s], kk, hf);
#pragma unroll
        for (int s = 0; s < 2; ++s) load_bfrag(b[s], wrow[s] + HH, kk, hf);
#pragma unroll
        for (int s = 0; s < 4; ++s)
#pragma unroll
            for (int s2 = 0; s2 < 2; ++s2)
                acc[s][s2] = __builtin_amdgcn_wmma_f32_16x16x32_bf16(
                    false, a[s].v, false, b[s2].v, (short)0, acc[s][s2], false, false);
    }
#pragma unroll
    for (int s = 0; s < 4; ++s)
#pragma unroll
        for (int s2 = 0; s2 < 2; ++s2)
#pragma unroll
            for (int r = 0; r < 8; ++r)
                go[(tm * 64 + s * 16 + r + hf * 8) * GG4 + col[s2]] = acc[s][s2][r];
}

// ---------------- LSTM gates + state update ----------------
__global__ void lstm_gates(const float* __restrict__ g,
                           const float* __restrict__ biasf, const float* __restrict__ biasb,
                           float* __restrict__ hsf, float* __restrict__ csf,
                           float* __restrict__ hsb, float* __restrict__ csb,
                           unsigned short* __restrict__ hstep16, int t)
{
    int i = blockIdx.x * blockDim.x + threadIdx.x;
    if (i >= 2 * NB * HH) return;
    int d = i % HH, b = (i / HH) % NB, dir = i / (NB * HH);
    const float* gg   = g + dir * (NB * GG4) + b * GG4;
    const float* bias = dir ? biasb : biasf;
    float gi = gg[d]          + bias[d];
    float gf = gg[d + HH]     + bias[d + HH];
    float gu = gg[d + 2 * HH] + bias[d + 2 * HH];
    float gn = gg[d + 3 * HH] + bias[d + 3 * HH];
    float* hs = dir ? hsb : hsf;
    float* cs = dir ? csb : csf;
    float cprev = (t > 0) ? cs[(b * LSEQ + t - 1) * HH + d] : 0.0f;
    float c  = sigf(gf) * cprev + sigf(gi) * tanhf(gu);
    float hv = sigf(gn) * tanhf(c);
    hs[(b * LSEQ + t) * HH + d] = hv;
    cs[(b * LSEQ + t) * HH + d] = c;
    hstep16[dir * (NB * HH) + b * HH + d] = f2bf(hv);
}

// ---------------- build initial tree state (concat fwd + reversed bwd) ----------------
__global__ void tree_init(const float* __restrict__ hsf, const float* __restrict__ csf,
                          const float* __restrict__ hsb, const float* __restrict__ csb,
                          const int* __restrict__ len,
                          float* __restrict__ h, float* __restrict__ c,
                          unsigned short* __restrict__ h16)
{
    int i = blockIdx.x * blockDim.x + threadIdx.x;
    if (i >= NB * LSEQ * DD2) return;
    int d = i % DD2, t = (i / DD2) % LSEQ, b = i / (DD2 * LSEQ);
    float hv, cv;
    if (d < HH) {
        hv = hsf[(b * LSEQ + t) * HH + d];
        cv = csf[(b * LSEQ + t) * HH + d];
    } else {
        int ln = len[b];
        int rt = (t < ln) ? (ln - 1 - t) : t;
        hv = hsb[(b * LSEQ + rt) * HH + (d - HH)];
        cv = csb[(b * LSEQ + rt) * HH + (d - HH)];
    }
    h[i] = hv; c[i] = cv; h16[i] = f2bf(hv);
}

// ---------------- tree GEMM: v = [h_l, h_r] @ w_comp^T ----------------
// Register-blocked 64x32 macro-tile per wave (4x2 WMMA tiles).
__global__ void tree_gemm(const unsigned short* __restrict__ h16,
                          const unsigned short* __restrict__ wcomp16,
                          float* __restrict__ v, int m)
{
    const int wave = threadIdx.x >> 5;
    const int lane = threadIdx.x & 31;
    const int idx  = blockIdx.x * 4 + wave;
    const int NMAC = (NCC / 16) / 2;           // 160 macro-cols
    const int MMAC = (NB * m / 16) / 4;        // 4m macro-rows
    if (idx >= MMAC * NMAC) return;            // wave-uniform guard
    const int tm = idx / NMAC, tn = idx % NMAC;
    const int lr = lane & 15, hf = lane >> 4;

    const unsigned short* hl[4];
#pragma unroll
    for (int s = 0; s < 4; ++s) {
        int r  = tm * 64 + s * 16 + lr;
        int bb = r / m, j = r - bb * m;
        hl[s] = h16 + (bb * LSEQ + j) * DD2;   // [h_l | h_r] contiguous (h_r = hl + DD2)
    }
    const unsigned short* wrow[2];
    int col[2];
#pragma unroll
    for (int s = 0; s < 2; ++s) {
        col[s]  = tn * 32 + s * 16 + lr;
        wrow[s] = wcomp16 + col[s] * KCC;
    }

    v8f acc[4][2] = {};
#pragma unroll 2
    for (int kk = 0; kk < KCC; kk += 32) {
        Frag16 a[4], b[2];
#pragma unroll
        for (int s = 0; s < 4; ++s) load_afrag(a[s], hl[s], kk, hf);
#pragma unroll
        for (int s = 0; s < 2; ++s) load_bfrag(b[s], wrow[s], kk, hf);
#pragma unroll
        for (int s = 0; s < 4; ++s)
#pragma unroll
            for (int s2 = 0; s2 < 2; ++s2)
                acc[s][s2] = __builtin_amdgcn_wmma_f32_16x16x32_bf16(
                    false, a[s].v, false, b[s2].v, (short)0, acc[s][s2], false, false);
    }
#pragma unroll
    for (int s = 0; s < 4; ++s)
#pragma unroll
        for (int s2 = 0; s2 < 2; ++s2)
#pragma unroll
            for (int rr = 0; rr < 8; ++rr)
                v[(tm * 64 + s * 16 + rr + hf * 8) * NCC + col[s2]] = acc[s][s2][rr];
}

// ---------------- tree gates ----------------
__global__ void tree_gates(const float* __restrict__ v, const float* __restrict__ bcomp,
                           const float* __restrict__ c,
                           float* __restrict__ nh, float* __restrict__ nc, int m)
{
    int i = blockIdx.x * blockDim.x + threadIdx.x;
    if (i >= NB * m * DD2) return;
    int d = i % DD2, j = (i / DD2) % m, b = i / (DD2 * m);
    const float* vr = v + (b * m + j) * NCC;
    float gi = vr[d]           + bcomp[d];
    float fl = vr[d + DD2]     + bcomp[d + DD2];
    float fr = vr[d + 2 * DD2] + bcomp[d + 2 * DD2];
    float gu = vr[d + 3 * DD2] + bcomp[d + 3 * DD2];
    float gn = vr[d + 4 * DD2] + bcomp[d + 4 * DD2];
    float cl = c[(b * LSEQ + j) * DD2 + d];
    float cr = c[(b * LSEQ + j + 1) * DD2 + d];
    float ncv = cl * sigf(fl + 1.0f) + cr * sigf(fr + 1.0f) + tanhf(gu) * sigf(gi);
    float nhv = sigf(gn) * tanhf(ncv);
    nh[(b * m + j) * DD2 + d] = nhv;
    nc[(b * m + j) * DD2 + d] = ncv;
}

// ---------------- select (argmax over query logits) + merge ----------------
__global__ void tree_select(const float* __restrict__ nh, const float* __restrict__ nc,
                            const float* __restrict__ hsrc, const float* __restrict__ csrc,
                            const float* __restrict__ wq, const int* __restrict__ len,
                            float* __restrict__ hdst, float* __restrict__ cdst,
                            unsigned short* __restrict__ h16dst,
                            int m, int it, int has_select)
{
    __shared__ float red[256];
    __shared__ float logits[16];
    __shared__ int bestS;
    const int b = blockIdx.x, tid = threadIdx.x;
    const int ln = len[b];
    const int done = ((it + 1) < ln) ? 1 : 0;
    int best = 0;

    if (has_select) {
        for (int j = 0; j < m; ++j) {
            float p = 0.0f;
            for (int d = tid; d < DD2; d += 256)
                p += nh[(b * m + j) * DD2 + d] * wq[d];
            red[tid] = p;
            __syncthreads();
            for (int s = 128; s > 0; s >>= 1) {
                if (tid < s) red[tid] += red[tid + s];
                __syncthreads();
            }
            if (tid == 0) logits[j] = red[0];
            __syncthreads();
        }
        if (tid == 0) {
            float mb = -3.4e38f; int bi = 0;
            for (int j = 0; j < m; ++j) {
                float mv = logits[j] + (((it + 1 + j) < ln) ? 0.0f : -10000.0f);
                if (mv > mb) { mb = mv; bi = j; }     // first max, like jnp.argmax
            }
            bestS = bi;
        }
        __syncthreads();
        best = bestS;
    }

    for (int idx = tid; idx < m * DD2; idx += 256) {
        int j = idx / DD2, d = idx - j * DD2;
        float hv, cv;
        if (!done) {
            hv = hsrc[(b * LSEQ + j) * DD2 + d];
            cv = csrc[(b * LSEQ + j) * DD2 + d];
        } else if (has_select && j < best) {
            hv = hsrc[(b * LSEQ + j) * DD2 + d];
            cv = csrc[(b * LSEQ + j) * DD2 + d];
        } else if (has_select && j > best) {
            hv = hsrc[(b * LSEQ + j + 1) * DD2 + d];
            cv = csrc[(b * LSEQ + j + 1) * DD2 + d];
        } else {
            hv = nh[(b * m + j) * DD2 + d];
            cv = nc[(b * m + j) * DD2 + d];
        }
        hdst[(b * LSEQ + j) * DD2 + d] = hv;
        cdst[(b * LSEQ + j) * DD2 + d] = cv;
        h16dst[(b * LSEQ + j) * DD2 + d] = f2bf(hv);
    }
}

__global__ void final_copy(const float* __restrict__ hfin, float* __restrict__ out)
{
    int i = blockIdx.x * blockDim.x + threadIdx.x;
    if (i >= NB * DD2) return;
    int b = i / DD2, d = i % DD2;
    out[i] = hfin[(b * LSEQ) * DD2 + d];
}

// ---------------- host launch ----------------
extern "C" void kernel_launch(void* const* d_in, const int* in_sizes, int n_in,
                              void* d_out, int out_size, void* d_ws, size_t ws_size,
                              hipStream_t stream)
{
    (void)in_sizes; (void)n_in; (void)out_size; (void)ws_size;
    const float* x     = (const float*)d_in[0];
    const int*   len   = (const int*)d_in[1];
    const float* wihf  = (const float*)d_in[2];
    const float* whhf  = (const float*)d_in[3];
    const float* biasf = (const float*)d_in[4];
    const float* wihb  = (const float*)d_in[5];
    const float* whhb  = (const float*)d_in[6];
    const float* biasb = (const float*)d_in[7];
    const float* wcomp = (const float*)d_in[8];
    const float* bcomp = (const float*)d_in[9];
    const float* wq    = (const float*)d_in[10];
    float* out = (float*)d_out;

    char* ws = (char*)d_ws;
    size_t off = 0;
    auto alloc = [&](size_t bytes) -> char* {
        char* p = ws + off;
        off = (off + bytes + 255) & ~(size_t)255;
        return p;
    };
    unsigned short* x16f    = (unsigned short*)alloc((size_t)NB * LSEQ * HH * 2);
    unsigned short* x16b    = (unsigned short*)alloc((size_t)NB * LSEQ * HH * 2);
    unsigned short* wcatf   = (unsigned short*)alloc((size_t)GG4 * KLL * 2);
    unsigned short* wcatb   = (unsigned short*)alloc((size_t)GG4 * KLL * 2);
    unsigned short* wcomp16 = (unsigned short*)alloc((size_t)NCC * KCC * 2);
    unsigned short* hstep16 = (unsigned short*)alloc((size_t)2 * NB * HH * 2);
    float* g   = (float*)alloc((size_t)2 * NB * GG4 * 4);
    float* hsf = (float*)alloc((size_t)NB * LSEQ * HH * 4);
    float* csf = (float*)alloc((size_t)NB * LSEQ * HH * 4);
    float* hsb = (float*)alloc((size_t)NB * LSEQ * HH * 4);
    float* csb = (float*)alloc((size_t)NB * LSEQ * HH * 4);
    float* hA  = (float*)alloc((size_t)NB * LSEQ * DD2 * 4);
    float* cA  = (float*)alloc((size_t)NB * LSEQ * DD2 * 4);
    float* hB  = (float*)alloc((size_t)NB * LSEQ * DD2 * 4);
    float* cB  = (float*)alloc((size_t)NB * LSEQ * DD2 * 4);
    unsigned short* h16A = (unsigned short*)alloc((size_t)NB * LSEQ * DD2 * 2);
    unsigned short* h16B = (unsigned short*)alloc((size_t)NB * LSEQ * DD2 * 2);
    float* vbuf = (float*)alloc((size_t)NB * 15 * NCC * 4);
    float* nh   = (float*)alloc((size_t)NB * 15 * DD2 * 4);
    float* nc   = (float*)alloc((size_t)NB * 15 * DD2 * 4);

    const int TPB = 256;
    // --- prep ---
    prep_x<<<dim3((NB * LSEQ * HH + TPB - 1) / TPB), TPB, 0, stream>>>(x, len, x16f, x16b);
    prep_wcat<<<dim3((GG4 * KLL + TPB - 1) / TPB), TPB, 0, stream>>>(wihf, whhf, wcatf);
    prep_wcat<<<dim3((GG4 * KLL + TPB - 1) / TPB), TPB, 0, stream>>>(wihb, whhb, wcatb);
    prep_bf16<<<dim3((NCC * KCC + TPB - 1) / TPB), TPB, 0, stream>>>(wcomp, wcomp16, NCC * KCC);
    zero_u16<<<dim3((2 * NB * HH + TPB - 1) / TPB), TPB, 0, stream>>>(hstep16, 2 * NB * HH);

    // --- BiLSTM scan: 16 steps, both directions per launch ---
    const int lstmMacro = ((NB / 16) / 4) * ((GG4 / 16) / 2);   // 256 macro-tiles
    for (int t = 0; t < LSEQ; ++t) {
        lstm_step_gemm<<<dim3((lstmMacro + 3) / 4, 2), 128, 0, stream>>>(
            x16f, x16b, wcatf, wcatb, hstep16, g, t);
        lstm_gates<<<dim3((2 * NB * HH + TPB - 1) / TPB), TPB, 0, stream>>>(
            g, biasf, biasb, hsf, csf, hsb, csb, hstep16, t);
    }

    // --- initial tree state ---
    tree_init<<<dim3((NB * LSEQ * DD2 + TPB - 1) / TPB), TPB, 0, stream>>>(
        hsf, csf, hsb, csb, len, hA, cA, h16A);

    // --- tree composition loop ---
    float* hcur = hA; float* ccur = cA; unsigned short* h16cur = h16A;
    float* hnx  = hB; float* cnx  = cB; unsigned short* h16nx  = h16B;
    for (int i = 0; i < LSEQ - 1; ++i) {
        int m = (LSEQ - 1) - i;
        int macro = ((NB * m / 16) / 4) * ((NCC / 16) / 2);     // 640*m macro-tiles
        tree_gemm<<<dim3((macro + 3) / 4), 128, 0, stream>>>(h16cur, wcomp16, vbuf, m);
        tree_gates<<<dim3((NB * m * DD2 + TPB - 1) / TPB), TPB, 0, stream>>>(
            vbuf, bcomp, ccur, nh, nc, m);
        int has_select = (i < LSEQ - 2) ? 1 : 0;
        tree_select<<<dim3(NB), TPB, 0, stream>>>(
            nh, nc, hcur, ccur, wq, len, hnx, cnx, h16nx, m, i, has_select);
        // ping-pong
        float* th = hcur; hcur = hnx; hnx = th;
        float* tc = ccur; ccur = cnx; cnx = tc;
        unsigned short* t16 = h16cur; h16cur = h16nx; h16nx = t16;
    }

    final_copy<<<dim3((NB * DD2 + TPB - 1) / TPB), TPB, 0, stream>>>(hcur, out);
}
